// MultiHeadSpatialAttention_27453430956279
// MI455X (gfx1250) — matrix operand order repaired and checked by reading, not compiled, not run on hardware
//
#include <hip/hip_runtime.h>
#include <hip/hip_bf16.h>
#include <math.h>

typedef __attribute__((ext_vector_type(16))) _Float16 v16h;
typedef __attribute__((ext_vector_type(8)))  _Float16 v8h;
typedef __attribute__((ext_vector_type(8)))  float    v8f;

#define D_MODEL   256
#define NUM_HEADS 8
#define DEPTH     32
#define SEQ       2048      // both Sq and S (CITY_SEQ)
#define BATCH     4
#define EPSW      1e-9f
#define CHUNK     64        // keys per LDS distance tile
#define PSTR      40        // padded LDS row stride (halves) for P staging

// ---------------------------------------------------------------------------
// WMMA helper: D = A(16x32 f16) * B(32x16 f16) + C(16x16 f32)
// ---------------------------------------------------------------------------
__device__ __forceinline__ v8f wmma_f16(v16h a, v16h b, v8f c) {
  return __builtin_amdgcn_wmma_f32_16x16x32_f16(false, a, false, b, (short)0, c,
                                                false, false);
}

// ---------------------------------------------------------------------------
// Projection GEMM:  out[m,n] = (sum_k A[m,k]*W[n,k] + bias[n]) * scale  -> f16
// A is f32 [M,256], W is f32 [256,256] (row n = output feature).
// STORE_MODE 0: row-major f16 [M,256]
// STORE_MODE 1: depth-major   f16 [b][h][dcol][s]   (for V: contiguous keys)
// Block: 256 thr = 8 waves; wave w owns N-tile (blockIdx.y*8 + w).
// ---------------------------------------------------------------------------
template <int STORE_MODE>
__global__ __launch_bounds__(256) void proj_f32_kernel(
    const float* __restrict__ A, const float* __restrict__ W,
    const float* __restrict__ bias, _Float16* __restrict__ out, float scale) {
  const int lane = threadIdx.x & 31;
  const int wave = threadIdx.x >> 5;
  const int half = lane >> 4;
  const int l16  = lane & 15;
  const int m0   = blockIdx.x * 16;
  const int n0   = (blockIdx.y * 8 + wave) * 16;
  const int nabs = n0 + l16;

  const float* Arow = A + (size_t)(m0 + l16) * D_MODEL;  // A: lane row = l16
  const float* Wrow = W + (size_t)nabs * D_MODEL;        // B: lane col = nabs

  v8f c = {};
#pragma unroll
  for (int ks = 0; ks < D_MODEL; ks += 32) {
    v16h a, b;
    const float* p1 = Arow + ks + 8 * half;        // K = 8*half + e
    const float* p2 = Arow + ks + 16 + 8 * half;   // K = 16 + 8*half + e
#pragma unroll
    for (int e = 0; e < 8; ++e) { a[e] = (_Float16)p1[e]; a[8 + e] = (_Float16)p2[e]; }
    const float* pw = Wrow + ks + 8 * half;        // K = 8*half + e, 16 + 8*half + e
#pragma unroll
    for (int e = 0; e < 16; ++e) b[e] = (_Float16)pw[e < 8 ? e : e + 8];
    c = wmma_f16(a, b, c);
  }

  const float bv = bias[nabs];
#pragma unroll
  for (int r = 0; r < 8; ++r) {
    const int m = m0 + r + 8 * half;               // C: row = r + 8*half
    const _Float16 hv = (_Float16)((c[r] + bv) * scale);
    if (STORE_MODE == 0) {
      out[(size_t)m * D_MODEL + nabs] = hv;
    } else {
      const int bidx = m / SEQ, s = m % SEQ;
      const int h = nabs >> 5, dcol = nabs & 31;
      out[(((size_t)bidx * NUM_HEADS + h) * DEPTH + dcol) * SEQ + s] = hv;
    }
  }
}

// ---------------------------------------------------------------------------
// Row-sums of inverse distance: rs[row] = sum_k 1/(dist[row,k]+eps)
// One wave per row.
// ---------------------------------------------------------------------------
__global__ __launch_bounds__(256) void rowsum_inv_kernel(
    const float* __restrict__ dist, float* __restrict__ rs) {
  const int lane = threadIdx.x & 31;
  const int row  = blockIdx.x * 8 + (threadIdx.x >> 5);
  const float* p = dist + (size_t)row * SEQ;
  float acc = 0.f;
  for (int i = lane; i < SEQ; i += 32) acc += 1.0f / (p[i] + EPSW);
#pragma unroll
  for (int m = 16; m >= 1; m >>= 1) acc += __shfl_xor(acc, m, 32);
  if (lane == 0) rs[row] = acc;
}

// ---------------------------------------------------------------------------
// Fused distance-weighted flash attention.
// Grid: (Sq/16, B). Block: 256 thr = 8 waves, wave = head.
// LDS distance tile (16 q-rows x 64 keys) shared by all 8 heads.
// Online softmax in f32; P staged via LDS to reach the A-fragment swizzle.
// ---------------------------------------------------------------------------
__global__ __launch_bounds__(256) void attn_kernel(
    const _Float16* __restrict__ qh,    // [b*Sq][256] f16, pre-scaled 1/sqrt(d)
    const _Float16* __restrict__ kh,    // [s][256] f16
    const _Float16* __restrict__ vhT,   // [b][h][dcol][s] f16
    const float* __restrict__ dist,     // [b][Sq][S] f32
    const float* __restrict__ rowsum,   // [b*Sq] f32
    _Float16* __restrict__ attn_out) {  // [b*Sq][256] f16
  __shared__ float    sdist[16][CHUNK];
  __shared__ _Float16 sP[NUM_HEADS][16][PSTR];

  const int lane = threadIdx.x & 31;
  const int wave = threadIdx.x >> 5;   // head
  const int half = lane >> 4;
  const int l16  = lane & 15;
  const int q0   = blockIdx.x * 16;
  const int b    = blockIdx.y;
  const int h    = wave;

  // Q A-fragment (row = l16, swizzled K runs), loaded once.
  v16h aq;
  {
    const _Float16* qp =
        qh + ((size_t)(b * SEQ + q0 + l16)) * D_MODEL + h * DEPTH;
    const v8h r1 = *(const v8h*)(qp + 8 * half);
    const v8h r2 = *(const v8h*)(qp + 16 + 8 * half);
#pragma unroll
    for (int e = 0; e < 8; ++e) { aq[e] = r1[e]; aq[8 + e] = r2[e]; }
  }

  float irs[8], mrow[8], lrow[8];
#pragma unroll
  for (int r = 0; r < 8; ++r) {
    irs[r]  = 1.0f / rowsum[b * SEQ + q0 + r + 8 * half];
    mrow[r] = -__builtin_huge_valf();
    lrow[r] = 0.0f;
  }
  v8f o0 = {}, o1 = {};

  const float* distbase = dist + ((size_t)b * SEQ + q0) * SEQ;
  const _Float16* vbase = vhT + ((size_t)b * NUM_HEADS + h) * DEPTH * SEQ;

  for (int s0 = 0; s0 < SEQ; s0 += CHUNK) {
    // Cooperative distance tile load (coalesced along keys) + prefetch next.
    for (int i = threadIdx.x; i < 16 * CHUNK; i += 256) {
      const int rr = i >> 6, cc = i & 63;
      const float* gp = distbase + (size_t)rr * SEQ + s0 + cc;
      sdist[rr][cc] = *gp;
      if (s0 + CHUNK < SEQ) __builtin_prefetch(gp + CHUNK, 0, 1);
    }
    __syncthreads();

#pragma unroll
    for (int sub = 0; sub < 2; ++sub) {   // 32 keys per inner step
      const int sb = s0 + sub * 32;

      // ---- scores: two 16-key tiles --------------------------------------
      v8f c0 = {}, c1 = {};
      {
        const _Float16* kp0 =
            kh + (size_t)(sb + l16) * D_MODEL + h * DEPTH + 8 * half;
        const _Float16* kp1 =
            kh + (size_t)(sb + 16 + l16) * D_MODEL + h * DEPTH + 8 * half;
        const v8h k0a = *(const v8h*)kp0;  // K = 8*half + e, 16 + 8*half + e
        const v8h k0b = *(const v8h*)(kp0 + 16);
        const v8h k1a = *(const v8h*)kp1;
        const v8h k1b = *(const v8h*)(kp1 + 16);
        v16h bf0, bf1;
#pragma unroll
        for (int e = 0; e < 8; ++e) {
          bf0[e] = k0a[e]; bf0[8 + e] = k0b[e];
          bf1[e] = k1a[e]; bf1[8 + e] = k1b[e];
        }
        c0 = wmma_f16(aq, bf0, c0);
        c1 = wmma_f16(aq, bf1, c1);
      }

      // ---- IDW weighting + online softmax over this 32-key block ---------
#pragma unroll
      for (int r = 0; r < 8; ++r) {
        const int row = r + 8 * half;
        const float d0 = sdist[row][sub * 32 + l16];
        const float d1 = sdist[row][sub * 32 + 16 + l16];
        float p0 = c0[r] * (1.0f / (d0 + EPSW)) * irs[r];
        float p1 = c1[r] * (1.0f / (d1 + EPSW)) * irs[r];

        float mx = fmaxf(p0, p1);
#pragma unroll
        for (int msk = 1; msk <= 8; msk <<= 1)
          mx = fmaxf(mx, __shfl_xor(mx, msk, 32));
        const float mnew  = fmaxf(mrow[r], mx);
        const float alpha = __expf(mrow[r] - mnew);
        mrow[r] = mnew;

        p0 = __expf(p0 - mnew);
        p1 = __expf(p1 - mnew);
        float rsum = p0 + p1;
#pragma unroll
        for (int msk = 1; msk <= 8; msk <<= 1)
          rsum += __shfl_xor(rsum, msk, 32);
        lrow[r] = lrow[r] * alpha + rsum;
        o0[r] *= alpha;
        o1[r] *= alpha;

        sP[wave][row][l16]      = (_Float16)p0;   // P in (row, key) order
        sP[wave][row][16 + l16] = (_Float16)p1;
      }

      // ---- P A-fragment (re-swizzle via LDS) ------------------------------
      v16h pa;
      {
        const _Float16* pp = &sP[wave][l16][0];
#pragma unroll
        for (int e = 0; e < 16; ++e) {
          pa[e] = pp[16 * half + e];
        }
      }

      // ---- V B-fragments (depth-major: keys contiguous) -------------------
      const v16h vf0 = *(const v16h*)(vbase + (size_t)l16 * SEQ + sb + 16 * half);
      const v16h vf1 =
          *(const v16h*)(vbase + (size_t)(l16 + 16) * SEQ + sb + 16 * half);
      o0 = wmma_f16(pa, vf0, o0);
      o1 = wmma_f16(pa, vf1, o1);
    }
    __syncthreads();
  }

  // ---- normalize and store head output (f16, row-major [b*Sq][256]) ------
#pragma unroll
  for (int r = 0; r < 8; ++r) {
    const float invl = 1.0f / lrow[r];
    const int row = r + 8 * half;
    _Float16* op =
        attn_out + ((size_t)(b * SEQ + q0 + row)) * D_MODEL + h * DEPTH;
    op[l16]      = (_Float16)(o0[r] * invl);
    op[16 + l16] = (_Float16)(o1[r] * invl);
  }
}

// ---------------------------------------------------------------------------
// Final FC: out[m,n] = sum_k A16[m,k]*W[n,k] + bias[n]   (f16 in, f32 out)
// ---------------------------------------------------------------------------
__global__ __launch_bounds__(256) void fc_kernel(
    const _Float16* __restrict__ A, const float* __restrict__ W,
    const float* __restrict__ bias, float* __restrict__ out) {
  const int lane = threadIdx.x & 31;
  const int wave = threadIdx.x >> 5;
  const int half = lane >> 4;
  const int l16  = lane & 15;
  const int m0   = blockIdx.x * 16;
  const int nabs = (blockIdx.y * 8 + wave) * 16 + l16;

  const _Float16* Arow = A + (size_t)(m0 + l16) * D_MODEL;
  const float*    Wrow = W + (size_t)nabs * D_MODEL;

  v8f c = {};
#pragma unroll
  for (int ks = 0; ks < D_MODEL; ks += 32) {
    v16h a, b;
    const v8h r1 = *(const v8h*)(Arow + ks + 8 * half);
    const v8h r2 = *(const v8h*)(Arow + ks + 16 + 8 * half);
#pragma unroll
    for (int e = 0; e < 8; ++e) { a[e] = r1[e]; a[8 + e] = r2[e]; }
    const float* pw = Wrow + ks + 8 * half;
#pragma unroll
    for (int e = 0; e < 16; ++e) b[e] = (_Float16)pw[e < 8 ? e : e + 8];
    c = wmma_f16(a, b, c);
  }

  const float bv = bias[nabs];
#pragma unroll
  for (int r = 0; r < 8; ++r)
    out[(size_t)(m0 + r + 8 * half) * D_MODEL + nabs] = c[r] + bv;
}

// ---------------------------------------------------------------------------
extern "C" void kernel_launch(void* const* d_in, const int* in_sizes, int n_in,
                              void* d_out, int out_size, void* d_ws,
                              size_t ws_size, hipStream_t stream) {
  const float* query  = (const float*)d_in[0];
  const float* values = (const float*)d_in[1];
  const float* dist   = (const float*)d_in[2];
  const float* Wq_w   = (const float*)d_in[3];
  const float* Wq_b   = (const float*)d_in[4];
  const float* Wk_w   = (const float*)d_in[5];
  const float* Wk_b   = (const float*)d_in[6];
  const float* Wv_w   = (const float*)d_in[7];
  const float* Wv_b   = (const float*)d_in[8];
  const float* fc_w   = (const float*)d_in[9];
  const float* fc_b   = (const float*)d_in[10];
  const float* tkeys  = (const float*)d_in[11];
  float* out = (float*)d_out;

  const size_t MQ = (size_t)BATCH * SEQ;            // 8192 rows
  _Float16* qh   = (_Float16*)d_ws;                 // [8192][256]
  _Float16* kh   = qh + MQ * D_MODEL;               // [2048][256]
  _Float16* vhT  = kh + (size_t)SEQ * D_MODEL;      // [b][h][32][2048]
  float*    rsum = (float*)(vhT + MQ * D_MODEL);    // [8192]
  _Float16* aout = (_Float16*)(rsum + MQ);          // [8192][256]

  const float qscale = 0.1767766952966369f;         // 1/sqrt(DEPTH)
  dim3 blk(256);

  proj_f32_kernel<0><<<dim3(MQ / 16, 2), blk, 0, stream>>>(query, Wq_w, Wq_b,
                                                           qh, qscale);
  proj_f32_kernel<0><<<dim3(SEQ / 16, 2), blk, 0, stream>>>(tkeys, Wk_w, Wk_b,
                                                            kh, 1.0f);
  proj_f32_kernel<1><<<dim3(MQ / 16, 2), blk, 0, stream>>>(values, Wv_w, Wv_b,
                                                           vhT, 1.0f);
  rowsum_inv_kernel<<<dim3(MQ / 8), blk, 0, stream>>>(dist, rsum);
  attn_kernel<<<dim3(SEQ / 16, BATCH), blk, 0, stream>>>(qh, kh, vhT, dist,
                                                         rsum, aout);
  fc_kernel<<<dim3(MQ / 16, 2), blk, 0, stream>>>(aout, fc_w, fc_b, out);
}
